// VidHRFormerBlockEnc_39719857553698
// MI455X (gfx1250) — matrix operand order repaired and checked by reading, not compiled, not run on hardware
//
#include <hip/hip_runtime.h>

typedef _Float16 f16;
typedef _Float16 v8h  __attribute__((ext_vector_type(8)));
typedef _Float16 v16h __attribute__((ext_vector_type(16)));
typedef float    v8f  __attribute__((ext_vector_type(8)));

constexpr int N_ = 4, T_ = 16, H_ = 28, W_ = 28, C_ = 384;
constexpr int WIN_ = 7, HEADS_ = 12, DH_ = 32, HID_ = 1536, DFF_ = 1024;
constexpr int NT_ = N_ * T_;          // 64 frames
constexpr int HW_ = H_ * W_;          // 784
constexpr int M_  = NT_ * HW_;        // 50176 tokens
constexpr int S_  = WIN_ * WIN_;      // 49
constexpr int WPF_ = 16;              // 4x4 windows per frame
constexpr int BWIN_ = NT_ * WPF_;     // 1024 windows

__device__ __forceinline__ float gelu_erf(float x) {
    return 0.5f * x * (1.0f + erff(x * 0.70710678118654752440f));
}

// ---------------------------------------------------------------- converts
__global__ __launch_bounds__(256)
void k_f16cvt(const float* __restrict__ in, f16* __restrict__ out, int n) {
    int i = blockIdx.x * 256 + threadIdx.x;
    if (i < n) out[i] = (f16)in[i];
}

// x (N,T,C,H,W) -> xh (token=(f,hw), C)
__global__ __launch_bounds__(256)
void k_permute_in(const float* __restrict__ x, float* __restrict__ xh) {
    int idx = blockIdx.x * 256 + threadIdx.x;
    if (idx >= M_ * C_) return;
    int c = idx % C_;
    int token = idx / C_;
    int f = token / HW_, hw = token % HW_;
    xh[idx] = x[((size_t)f * C_ + c) * HW_ + hw];
}

// xh -> out (N,T,C,H,W)
__global__ __launch_bounds__(256)
void k_permute_out(const float* __restrict__ xh, float* __restrict__ out) {
    int idx = blockIdx.x * 256 + threadIdx.x;
    if (idx >= M_ * C_) return;
    int hw = idx % HW_;
    int r = idx / HW_;
    int c = r % C_, f = r / C_;
    out[idx] = xh[((size_t)f * HW_ + hw) * C_ + c];
}

// ------------------------------------------------------- token LayerNorms
// one wave per token (wave32), 12 channels/lane
__global__ __launch_bounds__(256)
void k_token_ln(const float* __restrict__ xin, const float* __restrict__ g,
                const float* __restrict__ b, f16* __restrict__ out0,
                f16* __restrict__ out1, const float* __restrict__ tpos) {
    int lane = threadIdx.x & 31;
    int token = blockIdx.x * 8 + (threadIdx.x >> 5);
    const float* xr = xin + (size_t)token * C_;
    float v[12];
    float s = 0.f, ss = 0.f;
#pragma unroll
    for (int i = 0; i < 12; ++i) {
        v[i] = xr[lane + i * 32];
        s += v[i]; ss += v[i] * v[i];
    }
#pragma unroll
    for (int o = 16; o > 0; o >>= 1) { s += __shfl_xor(s, o, 32); ss += __shfl_xor(ss, o, 32); }
    float mean = s * (1.0f / C_);
    float var  = ss * (1.0f / C_) - mean * mean;
    float rstd = rsqrtf(var + 1e-5f);
    int t = (token / HW_) % T_;
#pragma unroll
    for (int i = 0; i < 12; ++i) {
        int c = lane + i * 32;
        float y = (v[i] - mean) * rstd * g[c] + b[c];
        out0[(size_t)token * C_ + c] = (f16)y;
        if (out1) out1[(size_t)token * C_ + c] = (f16)(y + tpos[t * C_ + c]);
    }
}

// window-major LN: row r = (bwin, s); writes xwin, xwin+pos, and map[r]=token
__global__ __launch_bounds__(256)
void k_spatial_ln(const float* __restrict__ xin, const float* __restrict__ g,
                  const float* __restrict__ b, const float* __restrict__ pos,
                  f16* __restrict__ out_win, f16* __restrict__ out_qk,
                  int* __restrict__ map) {
    int lane = threadIdx.x & 31;
    int r = blockIdx.x * 8 + (threadIdx.x >> 5);
    int s = r % S_;
    int bwin = r / S_;
    int win = bwin % WPF_, f = bwin / WPF_;
    int qh = win / 4, qw = win % 4;
    int wi = s / WIN_, wj = s % WIN_;
    int token = f * HW_ + (qh * WIN_ + wi) * W_ + (qw * WIN_ + wj);
    if (lane == 0) map[r] = token;
    const float* xr = xin + (size_t)token * C_;
    float v[12];
    float sm = 0.f, ssq = 0.f;
#pragma unroll
    for (int i = 0; i < 12; ++i) {
        v[i] = xr[lane + i * 32];
        sm += v[i]; ssq += v[i] * v[i];
    }
#pragma unroll
    for (int o = 16; o > 0; o >>= 1) { sm += __shfl_xor(sm, o, 32); ssq += __shfl_xor(ssq, o, 32); }
    float mean = sm * (1.0f / C_);
    float var  = ssq * (1.0f / C_) - mean * mean;
    float rstd = rsqrtf(var + 1e-5f);
#pragma unroll
    for (int i = 0; i < 12; ++i) {
        int c = lane + i * 32;
        float y = (v[i] - mean) * rstd * g[c] + b[c];
        out_win[(size_t)r * C_ + c] = (f16)y;
        out_qk[(size_t)r * C_ + c]  = (f16)(y + pos[s * C_ + c]);
    }
}

// ------------------------------------------------------------- WMMA GEMM
// D = A(MxK, f16, ld=K) * Wt(NxK, f16)^T + bias ; block=256 (8 waves),
// block tile 128 rows x 64 cols, wave tile 16x64 (4 WMMA accumulators).
// EPI: 0 = bias + f16 store, 1 = bias + GELU + f16 store,
//      2 = bias + fp32 residual accumulate (optional row remap)
template <int EPI>
__global__ __launch_bounds__(256)
void k_gemm(const f16* __restrict__ A, const f16* __restrict__ Wt,
            const float* __restrict__ bias, f16* __restrict__ out16,
            float* __restrict__ resid, const int* __restrict__ row_map,
            int K, int ldOut) {
    const int lane = threadIdx.x & 31;
    const int wave = threadIdx.x >> 5;
    const int rowBase = blockIdx.x * 128 + wave * 16;
    const int colBase = blockIdx.y * 64;
    const int mn = lane & 15;   // row-in-tile for A / col-in-tile for B
    const int kh = lane >> 4;   // K-half selector

    v8f acc[4] = {};

    const f16* arow = A + (size_t)(rowBase + mn) * K + kh * 16;
    const f16* bcol = Wt + (size_t)(colBase + mn) * K + kh * 16;
    const size_t bstep = (size_t)16 * K;

    for (int kt = 0; kt < K; kt += 32) {
        v8h alo = *(const v8h*)(arow + kt);
        v8h ahi = *(const v8h*)(arow + kt + 8);
        v16h af = __builtin_shufflevector(alo, ahi,
                                          0, 1, 2, 3, 4, 5, 6, 7,
                                          8, 9, 10, 11, 12, 13, 14, 15);
        v16h bf0 = *(const v16h*)(bcol + kt);
        v16h bf1 = *(const v16h*)(bcol + bstep + kt);
        v16h bf2 = *(const v16h*)(bcol + 2 * bstep + kt);
        v16h bf3 = *(const v16h*)(bcol + 3 * bstep + kt);
        acc[0] = __builtin_amdgcn_wmma_f32_16x16x32_f16(false, af, false, bf0, (short)0, acc[0], false, false);
        acc[1] = __builtin_amdgcn_wmma_f32_16x16x32_f16(false, af, false, bf1, (short)0, acc[1], false, false);
        acc[2] = __builtin_amdgcn_wmma_f32_16x16x32_f16(false, af, false, bf2, (short)0, acc[2], false, false);
        acc[3] = __builtin_amdgcn_wmma_f32_16x16x32_f16(false, af, false, bf3, (short)0, acc[3], false, false);
    }

#pragma unroll
    for (int j = 0; j < 4; ++j) {
        int col = colBase + j * 16 + mn;
        float bb = bias[col];
#pragma unroll
        for (int r = 0; r < 8; ++r) {
            int row = rowBase + r + kh * 8;
            float v = acc[j][r] + bb;
            if (EPI == 1) v = gelu_erf(v);
            if (EPI == 0 || EPI == 1) {
                out16[(size_t)row * ldOut + col] = (f16)v;
            } else {
                int dr = row_map ? row_map[row] : row;
                resid[(size_t)dr * ldOut + col] += v;
            }
        }
    }
}

// ------------------------------------------------------ spatial attention
// one block per (window, head); S=49, dh=32
__global__ __launch_bounds__(64)
void k_attn_spatial(const f16* __restrict__ qk, const f16* __restrict__ v,
                    f16* __restrict__ out) {
    __shared__ float ks[S_][DH_];
    __shared__ float vs[S_][DH_];
    __shared__ float sc[S_][S_ + 1];
    int bwin = blockIdx.x, h = blockIdx.y, tid = threadIdx.x;
    for (int idx = tid; idx < S_ * DH_; idx += 64) {
        int s = idx >> 5, d = idx & 31;
        size_t row = (size_t)bwin * S_ + s;
        ks[s][d] = (float)qk[row * (2 * C_) + C_ + h * DH_ + d];
        vs[s][d] = (float)v[row * C_ + h * DH_ + d];
    }
    __syncthreads();
    if (tid < S_) {
        int s = tid;
        size_t row = (size_t)bwin * S_ + s;
        float q[DH_];
#pragma unroll
        for (int d = 0; d < DH_; ++d) q[d] = (float)qk[row * (2 * C_) + h * DH_ + d];
        float mx = -1e30f;
        for (int t2 = 0; t2 < S_; ++t2) {
            float dot = 0.f;
#pragma unroll
            for (int d = 0; d < DH_; ++d) dot += q[d] * ks[t2][d];
            dot *= 0.17677669529663687f;  // 1/sqrt(32)
            sc[s][t2] = dot;
            mx = fmaxf(mx, dot);
        }
        float sum = 0.f;
        for (int t2 = 0; t2 < S_; ++t2) {
            float e = expf(sc[s][t2] - mx);
            sc[s][t2] = e;
            sum += e;
        }
        float inv = 1.0f / sum;
        float o[DH_];
#pragma unroll
        for (int d = 0; d < DH_; ++d) o[d] = 0.f;
        for (int t2 = 0; t2 < S_; ++t2) {
            float p = sc[s][t2] * inv;
#pragma unroll
            for (int d = 0; d < DH_; ++d) o[d] += p * vs[t2][d];
        }
#pragma unroll
        for (int d = 0; d < DH_; ++d) out[row * C_ + h * DH_ + d] = (f16)o[d];
    }
}

// ---------------------------------------------------- temporal attention
// one block per (spatial site b=(n,hw), head); S=16, causal
__global__ __launch_bounds__(32)
void k_attn_temporal(const f16* __restrict__ qk, const f16* __restrict__ v,
                     f16* __restrict__ out) {
    __shared__ float ks[T_][DH_];
    __shared__ float vs[T_][DH_];
    int b = blockIdx.x, h = blockIdx.y, tid = threadIdx.x;
    int n = b / HW_, hw = b % HW_;
    for (int idx = tid; idx < T_ * DH_; idx += 32) {
        int t2 = idx >> 5, d = idx & 31;
        size_t row = (size_t)(n * T_ + t2) * HW_ + hw;
        ks[t2][d] = (float)qk[row * (2 * C_) + C_ + h * DH_ + d];
        vs[t2][d] = (float)v[row * C_ + h * DH_ + d];
    }
    __syncthreads();
    if (tid < T_) {
        int s = tid;
        size_t row = (size_t)(n * T_ + s) * HW_ + hw;
        float q[DH_];
#pragma unroll
        for (int d = 0; d < DH_; ++d) q[d] = (float)qk[row * (2 * C_) + h * DH_ + d];
        float scv[T_];
        float mx = -1e30f;
        for (int t2 = 0; t2 <= s; ++t2) {
            float dot = 0.f;
#pragma unroll
            for (int d = 0; d < DH_; ++d) dot += q[d] * ks[t2][d];
            dot *= 0.17677669529663687f;
            scv[t2] = dot;
            mx = fmaxf(mx, dot);
        }
        float sum = 0.f;
        for (int t2 = 0; t2 <= s; ++t2) {
            float e = expf(scv[t2] - mx);
            scv[t2] = e;
            sum += e;
        }
        float inv = 1.0f / sum;
        float o[DH_];
#pragma unroll
        for (int d = 0; d < DH_; ++d) o[d] = 0.f;
        for (int t2 = 0; t2 <= s; ++t2) {
            float p = scv[t2] * inv;
#pragma unroll
            for (int d = 0; d < DH_; ++d) o[d] += p * vs[t2][d];
        }
#pragma unroll
        for (int d = 0; d < DH_; ++d) out[row * C_ + h * DH_ + d] = (f16)o[d];
    }
}

// -------------------------------------- per-frame LN stats (deterministic)
__global__ __launch_bounds__(256)
void k_stats_part(const f16* __restrict__ x, int epf,
                  float* __restrict__ ps, float* __restrict__ pss) {
    int f = blockIdx.y, ck = blockIdx.x, tid = threadIdx.x;
    const f16* base = x + (size_t)f * epf;
    int start = ck * 4096;
    int end = start + 4096;
    if (end > epf) end = epf;
    float s = 0.f, ss = 0.f;
    for (int i = start + tid; i < end; i += 256) {
        float v = (float)base[i];
        s += v; ss += v * v;
    }
    __shared__ float sh1[256], sh2[256];
    sh1[tid] = s; sh2[tid] = ss;
    __syncthreads();
    for (int o = 128; o > 0; o >>= 1) {
        if (tid < o) { sh1[tid] += sh1[tid + o]; sh2[tid] += sh2[tid + o]; }
        __syncthreads();
    }
    if (tid == 0) {
        ps[f * gridDim.x + ck]  = sh1[0];
        pss[f * gridDim.x + ck] = sh2[0];
    }
}

__global__ __launch_bounds__(256)
void k_stats_final(const float* __restrict__ ps, const float* __restrict__ pss,
                   int chunks, float cnt, float* __restrict__ mean,
                   float* __restrict__ rstd) {
    int f = blockIdx.x, tid = threadIdx.x;
    float s = 0.f, ss = 0.f;
    for (int i = tid; i < chunks; i += 256) {
        s += ps[f * chunks + i];
        ss += pss[f * chunks + i];
    }
    __shared__ float sh1[256], sh2[256];
    sh1[tid] = s; sh2[tid] = ss;
    __syncthreads();
    for (int o = 128; o > 0; o >>= 1) {
        if (tid < o) { sh1[tid] += sh1[tid + o]; sh2[tid] += sh2[tid + o]; }
        __syncthreads();
    }
    if (tid == 0) {
        float m = sh1[0] / cnt;
        float v = sh2[0] / cnt - m * m;
        mean[f] = m;
        rstd[f] = rsqrtf(v + 1e-5f);
    }
}

// MODE 0: out16 = gelu(ln(x));  MODE 1: resid += gelu(ln(x))
template <int MODE>
__global__ __launch_bounds__(256)
void k_cln_apply(const f16* __restrict__ xin, const float* __restrict__ g,
                 const float* __restrict__ b, const float* __restrict__ mean,
                 const float* __restrict__ rstd, f16* __restrict__ out16,
                 float* __restrict__ resid, int Cdim, size_t total) {
    size_t idx = (size_t)blockIdx.x * 256 + threadIdx.x;
    if (idx >= total) return;
    size_t token = idx / Cdim;
    int ch = (int)(idx % Cdim);
    int f = (int)(token / HW_);
    int pos = (int)(token % HW_);
    float v = (float)xin[idx];
    v = (v - mean[f]) * rstd[f] * g[(size_t)pos * Cdim + ch] + b[(size_t)pos * Cdim + ch];
    v = gelu_erf(v);
    if (MODE == 0) out16[idx] = (f16)v;
    else resid[idx] += v;
}

// depthwise 3x3 SAME conv per frame
__global__ __launch_bounds__(256)
void k_dwconv(const f16* __restrict__ in, const float* __restrict__ wgt,
              const float* __restrict__ bias, f16* __restrict__ out) {
    size_t idx = (size_t)blockIdx.x * 256 + threadIdx.x;
    if (idx >= (size_t)M_ * HID_) return;
    int ch = (int)(idx % HID_);
    size_t token = idx / HID_;
    int f = (int)(token / HW_);
    int pos = (int)(token % HW_);
    int hh = pos / W_, ww = pos % W_;
    float acc = bias[ch];
#pragma unroll
    for (int dy = 0; dy < 3; ++dy) {
        int y = hh + dy - 1;
        if (y < 0 || y >= H_) continue;
#pragma unroll
        for (int dx = 0; dx < 3; ++dx) {
            int xc = ww + dx - 1;
            if (xc < 0 || xc >= W_) continue;
            acc += (float)in[((size_t)f * HW_ + y * W_ + xc) * HID_ + ch] *
                   wgt[(dy * 3 + dx) * HID_ + ch];
        }
    }
    out[idx] = (f16)acc;
}

// ---------------------------------------------------------------- driver
extern "C" void kernel_launch(void* const* d_in, const int* in_sizes, int n_in,
                              void* d_out, int out_size, void* d_ws, size_t ws_size,
                              hipStream_t stream) {
    (void)in_sizes; (void)n_in; (void)out_size; (void)ws_size;
    const float* x        = (const float*)d_in[0];
    const float* pos_sp   = (const float*)d_in[1];
    const float* tpos     = (const float*)d_in[2];
    const float* ln1_g    = (const float*)d_in[3];
    const float* ln1_b    = (const float*)d_in[4];
    const float* ln2_g    = (const float*)d_in[5];
    const float* ln2_b    = (const float*)d_in[6];
    const float* sa_w_in  = (const float*)d_in[7];
    const float* sa_b_in  = (const float*)d_in[8];
    const float* sa_w_out = (const float*)d_in[9];
    const float* sa_b_out = (const float*)d_in[10];
    const float* fc1_w    = (const float*)d_in[11];
    const float* fc1_b    = (const float*)d_in[12];
    const float* cln1_w   = (const float*)d_in[13];
    const float* cln1_b   = (const float*)d_in[14];
    const float* dw_w     = (const float*)d_in[15];
    const float* dw_b     = (const float*)d_in[16];
    const float* cln2_w   = (const float*)d_in[17];
    const float* cln2_b   = (const float*)d_in[18];
    const float* fc2_w    = (const float*)d_in[19];
    const float* fc2_b    = (const float*)d_in[20];
    const float* cln3_w   = (const float*)d_in[21];
    const float* cln3_b   = (const float*)d_in[22];
    const float* ln3_g    = (const float*)d_in[23];
    const float* ln3_b    = (const float*)d_in[24];
    const float* ta_w_in  = (const float*)d_in[25];
    const float* ta_b_in  = (const float*)d_in[26];
    const float* ta_w_out = (const float*)d_in[27];
    const float* ta_b_out = (const float*)d_in[28];
    const float* ln4_g    = (const float*)d_in[29];
    const float* ln4_b    = (const float*)d_in[30];
    const float* lin1_w   = (const float*)d_in[31];
    const float* lin1_b   = (const float*)d_in[32];
    const float* lin2_w   = (const float*)d_in[33];
    const float* lin2_b   = (const float*)d_in[34];
    float* out = (float*)d_out;

    char* ws = (char*)d_ws;
    size_t off = 0;
    auto alloc = [&](size_t bytes) -> char* {
        char* p = ws + off;
        off += (bytes + 255) & ~(size_t)255;
        return p;
    };
    float* xh  = (float*)alloc((size_t)M_ * C_ * 4);
    f16* bufA  = (f16*)alloc((size_t)M_ * HID_ * 2);
    f16* bufB  = (f16*)alloc((size_t)M_ * HID_ * 2);
    f16* bufC  = (f16*)alloc((size_t)M_ * 768 * 2);
    f16* bufD  = (f16*)alloc((size_t)M_ * C_ * 2);
    f16* wf    = (f16*)alloc((size_t)3145728 * 2);
    int* map   = (int*)alloc((size_t)M_ * 4);
    float* ps  = (float*)alloc((size_t)NT_ * 294 * 4);
    float* pss = (float*)alloc((size_t)NT_ * 294 * 4);
    float* mb  = (float*)alloc(NT_ * 4);
    float* rb  = (float*)alloc(NT_ * 4);

    f16* wf_sa_in  = wf;
    f16* wf_sa_out = wf + 442368;
    f16* wf_fc1    = wf + 589824;
    f16* wf_fc2    = wf + 1179648;
    f16* wf_ta_in  = wf + 1769472;
    f16* wf_ta_out = wf + 2211840;
    f16* wf_lin1   = wf + 2359296;
    f16* wf_lin2   = wf + 2752512;

    auto cvt = [&](const float* src, f16* dst, int n) {
        k_f16cvt<<<(n + 255) / 256, 256, 0, stream>>>(src, dst, n);
    };
    cvt(sa_w_in, wf_sa_in, 1152 * 384);
    cvt(sa_w_out, wf_sa_out, 384 * 384);
    cvt(fc1_w, wf_fc1, 1536 * 384);
    cvt(fc2_w, wf_fc2, 384 * 1536);
    cvt(ta_w_in, wf_ta_in, 1152 * 384);
    cvt(ta_w_out, wf_ta_out, 384 * 384);
    cvt(lin1_w, wf_lin1, 1024 * 384);
    cvt(lin2_w, wf_lin2, 384 * 1024);

    k_permute_in<<<(M_ * C_ + 255) / 256, 256, 0, stream>>>(x, xh);

    // ---------------- spatial window attention ----------------
    k_spatial_ln<<<M_ / 8, 256, 0, stream>>>(xh, ln1_g, ln1_b, pos_sp, bufA, bufB, map);
    {   // Q,K projection (from xp+pos)
        dim3 g(M_ / 128, 768 / 64);
        k_gemm<0><<<g, 256, 0, stream>>>(bufB, wf_sa_in, sa_b_in, bufC, nullptr, nullptr, 384, 768);
    }
    {   // V projection (from xp)
        dim3 g(M_ / 128, 384 / 64);
        k_gemm<0><<<g, 256, 0, stream>>>(bufA, wf_sa_in + (size_t)768 * 384, sa_b_in + 768,
                                         bufD, nullptr, nullptr, 384, 384);
    }
    {
        dim3 g(BWIN_, HEADS_);
        k_attn_spatial<<<g, 64, 0, stream>>>(bufC, bufD, bufA);
    }
    {   // out-proj + residual (window un-permute via map)
        dim3 g(M_ / 128, 384 / 64);
        k_gemm<2><<<g, 256, 0, stream>>>(bufA, wf_sa_out, sa_b_out, nullptr, xh, map, 384, 384);
    }

    // ---------------- ConvFFN ----------------
    k_token_ln<<<M_ / 8, 256, 0, stream>>>(xh, ln2_g, ln2_b, bufD, nullptr, nullptr);
    {
        dim3 g(M_ / 128, 1536 / 64);
        k_gemm<0><<<g, 256, 0, stream>>>(bufD, wf_fc1, fc1_b, bufA, nullptr, nullptr, 384, 1536);
    }
    { dim3 g(294, NT_); k_stats_part<<<g, 256, 0, stream>>>(bufA, HW_ * HID_, ps, pss); }
    k_stats_final<<<NT_, 256, 0, stream>>>(ps, pss, 294, (float)(HW_ * HID_), mb, rb);
    {
        size_t tot = (size_t)M_ * HID_;
        k_cln_apply<0><<<(unsigned)((tot + 255) / 256), 256, 0, stream>>>(
            bufA, cln1_w, cln1_b, mb, rb, bufB, nullptr, HID_, tot);
    }
    {
        size_t tot = (size_t)M_ * HID_;
        k_dwconv<<<(unsigned)((tot + 255) / 256), 256, 0, stream>>>(bufB, dw_w, dw_b, bufA);
    }
    { dim3 g(294, NT_); k_stats_part<<<g, 256, 0, stream>>>(bufA, HW_ * HID_, ps, pss); }
    k_stats_final<<<NT_, 256, 0, stream>>>(ps, pss, 294, (float)(HW_ * HID_), mb, rb);
    {
        size_t tot = (size_t)M_ * HID_;
        k_cln_apply<0><<<(unsigned)((tot + 255) / 256), 256, 0, stream>>>(
            bufA, cln2_w, cln2_b, mb, rb, bufB, nullptr, HID_, tot);
    }
    {
        dim3 g(M_ / 128, 384 / 64);
        k_gemm<0><<<g, 256, 0, stream>>>(bufB, wf_fc2, fc2_b, bufA, nullptr, nullptr, 1536, 384);
    }
    { dim3 g(74, NT_); k_stats_part<<<g, 256, 0, stream>>>(bufA, HW_ * C_, ps, pss); }
    k_stats_final<<<NT_, 256, 0, stream>>>(ps, pss, 74, (float)(HW_ * C_), mb, rb);
    {
        size_t tot = (size_t)M_ * C_;
        k_cln_apply<1><<<(unsigned)((tot + 255) / 256), 256, 0, stream>>>(
            bufA, cln3_w, cln3_b, mb, rb, nullptr, xh, C_, tot);
    }

    // ---------------- temporal causal attention ----------------
    k_token_ln<<<M_ / 8, 256, 0, stream>>>(xh, ln3_g, ln3_b, bufD, bufB, tpos);
    {
        dim3 g(M_ / 128, 768 / 64);
        k_gemm<0><<<g, 256, 0, stream>>>(bufB, wf_ta_in, ta_b_in, bufC, nullptr, nullptr, 384, 768);
    }
    {
        dim3 g(M_ / 128, 384 / 64);
        k_gemm<0><<<g, 256, 0, stream>>>(bufD, wf_ta_in + (size_t)768 * 384, ta_b_in + 768,
                                         bufA, nullptr, nullptr, 384, 384);
    }
    {
        dim3 g(N_ * HW_, HEADS_);
        k_attn_temporal<<<g, 32, 0, stream>>>(bufC, bufA, bufB);
    }
    {
        dim3 g(M_ / 128, 384 / 64);
        k_gemm<2><<<g, 256, 0, stream>>>(bufB, wf_ta_out, ta_b_out, nullptr, xh, nullptr, 384, 384);
    }

    // ---------------- final FFN ----------------
    k_token_ln<<<M_ / 8, 256, 0, stream>>>(xh, ln4_g, ln4_b, bufA, nullptr, nullptr);
    {
        dim3 g(M_ / 128, 1024 / 64);
        k_gemm<1><<<g, 256, 0, stream>>>(bufA, wf_lin1, lin1_b, bufB, nullptr, nullptr, 384, 1024);
    }
    {
        dim3 g(M_ / 128, 384 / 64);
        k_gemm<2><<<g, 256, 0, stream>>>(bufB, wf_lin2, lin2_b, nullptr, xh, nullptr, 1024, 384);
    }

    k_permute_out<<<(M_ * C_ + 255) / 256, 256, 0, stream>>>(xh, out);
}
